// SlotAttention_54537494724887
// MI455X (gfx1250) — matrix operand order repaired and checked by reading, not compiled, not run on hardware
//
#include <hip/hip_runtime.h>
#include <cstdint>
#include <cstddef>

typedef __attribute__((ext_vector_type(16))) _Float16 v16h;
typedef __attribute__((ext_vector_type(8)))  float    v8f;

namespace {

constexpr int   NB    = 8;            // batch
constexpr int   N     = 16384;        // H*W
constexpr int   BN    = NB * N;       // 131072 rows
constexpr float LNEPS = 1e-5f;
constexpr float ATTEPS = 1e-8f;
constexpr float SCALE  = 0.125f;      // 64^-0.5

// ---- workspace layout (float offsets) ----
constexpr size_t OFF_FEATN  = 0;          // BN*64 ; region reused after GEMMs:
constexpr size_t OFF_E      = 0;          //   N*64
constexpr size_t OFF_FTF    = 1048576;    //   BN*8
constexpr size_t OFF_FTB    = 2097152;    //   BN*8
constexpr size_t OFF_LOGITS = 3145728;    //   B*8*N
constexpr size_t OFF_KFB    = 8388608;    // BN*64
constexpr size_t OFF_VFB    = 16777216;   // BN*64
constexpr size_t OFF_XKBG   = 25165824;   // BN*64 (LN'd bg key input)
constexpr size_t OFF_XVBG   = 33554432;   // BN*64 (LN'd bg value input)
constexpr size_t OFF_SMALL  = 41943040;
constexpr size_t OFF_SLOTF  = OFF_SMALL + 0;     // 8*7*64
constexpr size_t OFF_SLOTB  = OFF_SMALL + 3584;  // 8*64
constexpr size_t OFF_POS    = OFF_SMALL + 4096;  // 8*7*2
constexpr size_t OFF_WQF    = OFF_SMALL + 4224;  // 8*7*64
constexpr size_t OFF_CFC    = OFF_SMALL + 7808;  // 56
constexpr size_t OFF_CVEC   = OFF_SMALL + 7872;  // 8*7*64
constexpr size_t OFF_WQB    = OFF_SMALL + 11456; // 8*64
constexpr size_t OFF_CBC    = OFF_SMALL + 11968; // 8
constexpr size_t OFF_ASUM   = OFF_SMALL + 11976; // 64   } contiguous 176
constexpr size_t OFF_AX     = OFF_SMALL + 12040; // 56   }
constexpr size_t OFF_AY     = OFF_SMALL + 12096; // 56   }
constexpr size_t OFF_INVA   = OFF_SMALL + 12152; // 64
constexpr size_t OFF_SFG    = OFF_SMALL + 12216; // 3584 } contiguous 4096
constexpr size_t OFF_SBG    = OFF_SMALL + 15800; // 512  }
constexpr size_t OFF_STEX   = OFF_SMALL + 16312; // 512

// ---- output layout (floats) ----
constexpr size_t OUT_SLOTS = 0;        // (B,8,64)  = 4096
constexpr size_t OUT_ATTN  = 4096;     // (B,8,N)   = 1048576
constexpr size_t OUT_POS   = 1052672;  // (B,7,2)   = 112
constexpr size_t OUT_TEX   = 1052784;  // (B,8,8)   = 512

__device__ __forceinline__ float waveSum(float v) {
#pragma unroll
  for (int m = 16; m; m >>= 1) v += __shfl_xor(v, m, 32);
  return v;
}

__device__ __forceinline__ float blockRed64(float v, float* red2) {
  float s = waveSum(v);
  __syncthreads();
  if ((threadIdx.x & 31) == 0) red2[threadIdx.x >> 5] = s;
  __syncthreads();
  return red2[0] + red2[1];
}

__device__ __forceinline__ float blockRed256(float v, float* red8) {
  float s = waveSum(v);
  __syncthreads();
  if ((threadIdx.x & 31) == 0) red8[threadIdx.x >> 5] = s;
  __syncthreads();
  float t = 0.f;
  if (threadIdx.x == 0) {
#pragma unroll
    for (int i = 0; i < 8; ++i) t += red8[i];
  }
  return t;
}

// -------- generic zero --------
__global__ void zero_kernel(float* __restrict__ p, int n) {
  int i = blockIdx.x * 256 + threadIdx.x;
  if (i < n) p[i] = 0.f;
}

// -------- row LayerNorm over C=64 (32 lanes/row, 8 rows/block) --------
__global__ void ln64_kernel(const float* __restrict__ X, const float* __restrict__ g,
                            const float* __restrict__ b, float* __restrict__ Y) {
  int row  = blockIdx.x * 8 + (threadIdx.x >> 5);
  int lane = threadIdx.x & 31;
  float2 x = *reinterpret_cast<const float2*>(X + (size_t)row * 64 + 2 * lane);
  float s  = waveSum(x.x + x.y);
  float s2 = waveSum(x.x * x.x + x.y * x.y);
  float m  = s * (1.f / 64.f);
  float inv = rsqrtf(s2 * (1.f / 64.f) - m * m + LNEPS);
  float2 gg = *reinterpret_cast<const float2*>(g + 2 * lane);
  float2 bb = *reinterpret_cast<const float2*>(b + 2 * lane);
  float2 y;
  y.x = (x.x - m) * inv * gg.x + bb.x;
  y.y = (x.y - m) * inv * gg.y + bb.y;
  *reinterpret_cast<float2*>(Y + (size_t)row * 64 + 2 * lane) = y;
}

// -------- WMMA GEMM: Y[r][o] = sum_c X[r][c] * W[o][c]   (R x 64 @ 64 x 64) --------
// One wave per 16-row strip; f16 inputs, f32 accumulation via v_wmma_f32_16x16x32_f16.
__global__ void gemm64_wmma_kernel(const float* __restrict__ X, const float* __restrict__ W,
                                   float* __restrict__ Y) {
  __shared__ __align__(32) _Float16 Wh[64 * 64];
  int tid = threadIdx.x;
  for (int i = tid; i < 4096; i += 256) Wh[i] = (_Float16)W[i];
  __syncthreads();
  int wave = tid >> 5, lane = tid & 31;
  int m = lane & 15, h = lane >> 4;
  size_t row0 = (size_t)(blockIdx.x * 8 + wave) * 16;
  const float* xr = X + (row0 + (size_t)m) * 64;
  v8f acc[4] = {v8f{}, v8f{}, v8f{}, v8f{}};
#pragma unroll
  for (int q = 0; q < 2; ++q) {
    // A tile 16x32 f16 (wave32 layout): lanes 0-15 rows, K split by lane-half
    v16h A;
    int base = q * 32 + h * 16;
#pragma unroll
    for (int j = 0; j < 8; ++j) A[j] = (_Float16)xr[base + j];
#pragma unroll
    for (int j = 0; j < 8; ++j) A[8 + j] = (_Float16)xr[base + 8 + j];
#pragma unroll
    for (int t = 0; t < 4; ++t) {
      // B tile 32x16: lane = output col, halves = 16 consecutive K at 32q+16h
      const v16h Bv = *reinterpret_cast<const v16h*>(&Wh[(t * 16 + m) * 64 + q * 32 + h * 16]);
      acc[t] = __builtin_amdgcn_wmma_f32_16x16x32_f16(false, A, false, Bv, (short)0, acc[t],
                                                      false, false);
    }
  }
  float* yr = Y + row0 * 64;
#pragma unroll
  for (int t = 0; t < 4; ++t)
#pragma unroll
    for (int r = 0; r < 8; ++r)
      yr[(size_t)(r + 8 * h) * 64 + t * 16 + m] = acc[t][r];
}

// -------- grid embedding, position-independent part: E[n][c] = gx*u[c] + gy*w2[c] --------
__global__ void gridE_kernel(const float* __restrict__ gew, float* __restrict__ E) {
  int i = blockIdx.x * 256 + threadIdx.x;  // N*64
  int n = i >> 6, c = i & 63;
  float gx = -1.f + (float)(n & 127) * (2.f / 127.f);
  float gy = -1.f + (float)(n >> 7) * (2.f / 127.f);
  float u  = gew[c * 4 + 0] - gew[c * 4 + 2];
  float w2 = gew[c * 4 + 1] - gew[c * 4 + 3];
  E[i] = gx * u + gy * w2;
}

// -------- in-place: X = LN(X + E[n] + geb) with given LN params --------
__global__ void addE_ln_kernel(float* __restrict__ Xio, const float* __restrict__ E,
                               const float* __restrict__ geb, const float* __restrict__ g,
                               const float* __restrict__ b) {
  int row  = blockIdx.x * 8 + (threadIdx.x >> 5);
  int lane = threadIdx.x & 31;
  int n = row & (N - 1);
  float2 x = *reinterpret_cast<const float2*>(Xio + (size_t)row * 64 + 2 * lane);
  float2 e = *reinterpret_cast<const float2*>(E + (size_t)n * 64 + 2 * lane);
  float2 gb = *reinterpret_cast<const float2*>(geb + 2 * lane);
  x.x += e.x + gb.x; x.y += e.y + gb.y;
  float s  = waveSum(x.x + x.y);
  float s2 = waveSum(x.x * x.x + x.y * x.y);
  float m  = s * (1.f / 64.f);
  float inv = rsqrtf(s2 * (1.f / 64.f) - m * m + LNEPS);
  float2 gg = *reinterpret_cast<const float2*>(g + 2 * lane);
  float2 bb = *reinterpret_cast<const float2*>(b + 2 * lane);
  float2 y;
  y.x = (x.x - m) * inv * gg.x + bb.x;
  y.y = (x.y - m) * inv * gg.y + bb.y;
  *reinterpret_cast<float2*>(Xio + (size_t)row * 64 + 2 * lane) = y;
}

// -------- texture precompute: ftf/ftb = LN_T(ft) @ texW^T + texB (T=8) --------
__global__ void tex_pre_kernel(const float* __restrict__ ft, const float* fg_g,
                               const float* fg_b, const float* fg_w, const float* fg_bias,
                               const float* bg_g, const float* bg_b, const float* bg_w,
                               const float* bg_bias, float* __restrict__ ftf,
                               float* __restrict__ ftb) {
  int row = blockIdx.x * 256 + threadIdx.x;  // BN
  const float* x = ft + (size_t)row * 8;
  float v[8], s = 0.f, s2 = 0.f;
#pragma unroll
  for (int t = 0; t < 8; ++t) { v[t] = x[t]; s += v[t]; s2 += v[t] * v[t]; }
  float m = s * 0.125f;
  float inv = rsqrtf(s2 * 0.125f - m * m + LNEPS);
  float lf[8], lb[8];
#pragma unroll
  for (int t = 0; t < 8; ++t) {
    float xn = (v[t] - m) * inv;
    lf[t] = xn * fg_g[t] + fg_b[t];
    lb[t] = xn * bg_g[t] + bg_b[t];
  }
#pragma unroll
  for (int o = 0; o < 8; ++o) {
    float a = fg_bias[o], c = bg_bias[o];
#pragma unroll
    for (int t = 0; t < 8; ++t) {
      a += lf[t] * fg_w[o * 8 + t];
      c += lb[t] * bg_w[o * 8 + t];
    }
    ftf[(size_t)row * 8 + o] = a;
    ftb[(size_t)row * 8 + o] = c;
  }
}

// -------- slot / position init --------
__global__ void init_kernel(const float* smu, const float* sls, const float* smub,
                            const float* slsb, const float* nf, const float* nbg,
                            const float* fpos, float* slotf, float* slotb, float* pos) {
  int i = blockIdx.x * 256 + threadIdx.x;
  if (i < 3584) { int c = i & 63; slotf[i] = smu[c] + expf(sls[c]) * nf[i]; }
  if (i < 512)  { int c = i & 63; slotb[i] = smub[c] + expf(slsb[c]) * nbg[i]; }
  if (i < 112)  { pos[i] = fpos[i % 14]; }
}

// -------- per-iteration slot prep: q, folded weight vectors, position constants --------
__global__ void slot_prep_kernel(const float* slotf, const float* slotb, const float* pos,
                                 const float* qlng, const float* qlnb, const float* qw,
                                 const float* qblng, const float* qblnb, const float* qbw,
                                 const float* mfgw, const float* mfgb, const float* mbgw,
                                 const float* mbgb, const float* gew, const float* geb,
                                 float* wqf, float* cf, float* cvec, float* wqb, float* cb) {
  __shared__ float xln[64], qv[64], red2[2];
  int blk = blockIdx.x, tid = threadIdx.x;
  bool isfg = blk < 56;
  int row = isfg ? blk : blk - 56;
  const float* slot = (isfg ? slotf : slotb) + row * 64;
  const float* lng = isfg ? qlng : qblng;
  const float* lnb = isfg ? qlnb : qblnb;
  const float* qwm = isfg ? qw : qbw;
  const float* mw  = isfg ? mfgw : mbgw;
  const float* mb  = isfg ? mfgb : mbgb;

  float x = slot[tid];
  float s  = blockRed64(x, red2);
  float s2 = blockRed64(x * x, red2);
  float m  = s * (1.f / 64.f);
  float inv = rsqrtf(s2 * (1.f / 64.f) - m * m + LNEPS);
  xln[tid] = (x - m) * inv * lng[tid] + lnb[tid];
  __syncthreads();
  float q = 0.f;
  for (int c = 0; c < 64; ++c) q += xln[c] * qwm[tid * 64 + c];
  qv[tid] = q;
  __syncthreads();
  float wq = 0.f;
  for (int o = 0; o < 64; ++o) wq += mw[o * 64 + tid] * qv[o];
  float ctot = blockRed64(mb[tid] * qv[tid], red2);
  if (isfg) {
    wqf[row * 64 + tid] = wq;
    if (tid == 0) cf[row] = ctot;
    float px = pos[row * 2], py = pos[row * 2 + 1];
    float u  = gew[tid * 4 + 0] - gew[tid * 4 + 2];
    float w2 = gew[tid * 4 + 1] - gew[tid * 4 + 3];
    cvec[row * 64 + tid] = geb[tid] - px * u - py * w2;
  } else {
    wqb[row * 64 + tid] = wq;
    if (tid == 0) cb[row] = ctot;
  }
}

// -------- logits for all (b,k,n); k==0 is background --------
__global__ void logits_kernel(const float* __restrict__ kfb, const float* __restrict__ E,
                              const float* __restrict__ cvec, const float* __restrict__ wqf,
                              const float* __restrict__ cf, const float* __restrict__ xkbg,
                              const float* __restrict__ wqb, const float* __restrict__ cb,
                              const float* __restrict__ mlng, const float* __restrict__ mlnb,
                              float* __restrict__ logits) {
  int row  = blockIdx.x * 8 + (threadIdx.x >> 5);  // b*8*N + k*N + n
  int lane = threadIdx.x & 31;
  int b = row >> 17;
  int rem = row & ((8 << 14) - 1);
  int k = rem >> 14;
  int n = rem & (N - 1);
  if (k == 0) {
    float2 x = *reinterpret_cast<const float2*>(xkbg + ((size_t)(b << 14) + n) * 64 + 2 * lane);
    float2 w = *reinterpret_cast<const float2*>(wqb + b * 64 + 2 * lane);
    float d = waveSum(x.x * w.x + x.y * w.y);
    if (lane == 0) logits[row] = (d + cb[b]) * SCALE;
  } else {
    int fk = b * 7 + (k - 1);
    const float* xr = kfb + ((size_t)(b << 14) + n) * 64 + 2 * lane;
    const float* er = E + (size_t)n * 64 + 2 * lane;
    float2 cv = *reinterpret_cast<const float2*>(cvec + fk * 64 + 2 * lane);
    float2 x;
    x.x = xr[0] + er[0] + cv.x;
    x.y = xr[1] + er[1] + cv.y;
    float s  = waveSum(x.x + x.y);
    float s2 = waveSum(x.x * x.x + x.y * x.y);
    float m  = s * (1.f / 64.f);
    float inv = rsqrtf(s2 * (1.f / 64.f) - m * m + LNEPS);
    float2 g  = *reinterpret_cast<const float2*>(mlng + 2 * lane);
    float2 bb = *reinterpret_cast<const float2*>(mlnb + 2 * lane);
    float2 w  = *reinterpret_cast<const float2*>(wqf + fk * 64 + 2 * lane);
    float d = waveSum(((x.x - m) * inv * g.x + bb.x) * w.x +
                      ((x.y - m) * inv * g.y + bb.y) * w.y);
    if (lane == 0) logits[row] = (d + cf[fk]) * SCALE;
  }
}

// -------- softmax over K per pixel + fused sums (attn mass, attn*grid) --------
__global__ void softmax_kernel(const float* __restrict__ logits, float* __restrict__ attn_out,
                               float* __restrict__ Asum, float* __restrict__ Ax,
                               float* __restrict__ Ay) {
  __shared__ float red8[8];
  int gid = blockIdx.x * 256 + threadIdx.x;
  int b = gid >> 14, n = gid & (N - 1);
  float l[8], mx = -1e30f;
#pragma unroll
  for (int k = 0; k < 8; ++k) {
    l[k] = logits[((size_t)(b * 8 + k) << 14) + n];
    mx = fmaxf(mx, l[k]);
  }
  float sum = 0.f;
#pragma unroll
  for (int k = 0; k < 8; ++k) { l[k] = expf(l[k] - mx); sum += l[k]; }
  float inv = 1.f / sum;
  float a[8];
#pragma unroll
  for (int k = 0; k < 8; ++k) {
    a[k] = l[k] * inv + ATTEPS;
    attn_out[((size_t)(b * 8 + k) << 14) + n] = a[k];
  }
  float gx = -1.f + (float)(n & 127) * (2.f / 127.f);
  float gy = -1.f + (float)(n >> 7) * (2.f / 127.f);
#pragma unroll
  for (int k = 0; k < 8; ++k) {
    float t = blockRed256(a[k], red8);
    if (threadIdx.x == 0) atomicAdd(&Asum[b * 8 + k], t);
  }
#pragma unroll
  for (int k = 1; k < 8; ++k) {
    float tx = blockRed256(a[k] * gx, red8);
    if (threadIdx.x == 0) atomicAdd(&Ax[b * 7 + k - 1], tx);
    float ty = blockRed256(a[k] * gy, red8);
    if (threadIdx.x == 0) atomicAdd(&Ay[b * 7 + k - 1], ty);
  }
}

// -------- finalize attn sums: inv-norms + fg positions --------
__global__ void post_attn_kernel(const float* Asum, const float* Ax, const float* Ay,
                                 float* invA, float* pos, float* pos_out) {
  int tid = threadIdx.x;
  invA[tid] = 1.f / Asum[tid];
  if (tid < 56) {
    int b = tid / 7, kk = tid % 7;
    float s = 1.f / Asum[b * 8 + 1 + kk];
    float px = Ax[tid] * s, py = Ay[tid] * s;
    pos[tid * 2] = px;  pos[tid * 2 + 1] = py;
    pos_out[tid * 2] = px;  pos_out[tid * 2 + 1] = py;
  }
}

// -------- foreground update accumulation: s = sum_n w_n * LN(vfb + E + cvec) --------
__global__ void upd_fg_kernel(const float* __restrict__ vfb, const float* __restrict__ E,
                              const float* __restrict__ cvec, const float* __restrict__ attn,
                              const float* __restrict__ invA, const float* __restrict__ mlng,
                              const float* __restrict__ mlnb, float* __restrict__ sfg) {
  __shared__ float sacc[8][64];
  int blk = blockIdx.x;
  int chunk = blk & 7;
  int fk = blk >> 3;          // 0..55
  int b = fk / 7, kk = fk % 7;
  int wave = threadIdx.x >> 5, lane = threadIdx.x & 31;
  float2 cv = *reinterpret_cast<const float2*>(cvec + fk * 64 + 2 * lane);
  float2 g  = *reinterpret_cast<const float2*>(mlng + 2 * lane);
  float2 bb = *reinterpret_cast<const float2*>(mlnb + 2 * lane);
  float inva = invA[b * 8 + 1 + kk];
  const float* vb = vfb + ((size_t)b << 14) * 64;
  const float* at = attn + ((size_t)(b * 8 + 1 + kk) << 14);
  float2 acc = {0.f, 0.f};
  int nbase = chunk * 2048;
  for (int i = 0; i < 256; ++i) {
    int n = nbase + i * 8 + wave;
    float w = at[n] * inva;
    const float* xr = vb + (size_t)n * 64 + 2 * lane;
    const float* er = E + (size_t)n * 64 + 2 * lane;
    float2 x;
    x.x = xr[0] + er[0] + cv.x;
    x.y = xr[1] + er[1] + cv.y;
    float s  = waveSum(x.x + x.y);
    float s2 = waveSum(x.x * x.x + x.y * x.y);
    float m  = s * (1.f / 64.f);
    float inv = rsqrtf(s2 * (1.f / 64.f) - m * m + LNEPS);
    acc.x += w * ((x.x - m) * inv * g.x + bb.x);
    acc.y += w * ((x.y - m) * inv * g.y + bb.y);
  }
  sacc[wave][2 * lane] = acc.x;
  sacc[wave][2 * lane + 1] = acc.y;
  __syncthreads();
  if (threadIdx.x < 64) {
    float s = 0.f;
#pragma unroll
    for (int w = 0; w < 8; ++w) s += sacc[w][threadIdx.x];
    atomicAdd(&sfg[fk * 64 + threadIdx.x], s);
  }
}

// -------- background update accumulation (inputs already LN'd) --------
__global__ void upd_bg_kernel(const float* __restrict__ xvbg, const float* __restrict__ attn,
                              const float* __restrict__ invA, float* __restrict__ sbg) {
  __shared__ float sacc[8][64];
  int blk = blockIdx.x;
  int chunk = blk & 7;
  int b = blk >> 3;
  int wave = threadIdx.x >> 5, lane = threadIdx.x & 31;
  float inva = invA[b * 8];
  const float* vb = xvbg + ((size_t)b << 14) * 64;
  const float* at = attn + ((size_t)(b * 8) << 14);
  float2 acc = {0.f, 0.f};
  int nbase = chunk * 2048;
  for (int i = 0; i < 256; ++i) {
    int n = nbase + i * 8 + wave;
    float w = at[n] * inva;
    float2 x = *reinterpret_cast<const float2*>(vb + (size_t)n * 64 + 2 * lane);
    acc.x += w * x.x;
    acc.y += w * x.y;
  }
  sacc[wave][2 * lane] = acc.x;
  sacc[wave][2 * lane + 1] = acc.y;
  __syncthreads();
  if (threadIdx.x < 64) {
    float s = 0.f;
#pragma unroll
    for (int w = 0; w < 8; ++w) s += sacc[w][threadIdx.x];
    atomicAdd(&sbg[b * 64 + threadIdx.x], s);
  }
}

// -------- GRU cell + residual MLP on the 64 slot rows --------
__global__ void gru_res_kernel(
    float* __restrict__ slotf, float* __restrict__ slotb, const float* __restrict__ sfg,
    const float* __restrict__ sbg, const float* mfgw, const float* mfgb, const float* mbgw,
    const float* mbgb, const float* wih_f, const float* whh_f, const float* bih_f,
    const float* bhh_f, const float* wih_b, const float* whh_b, const float* bih_b,
    const float* bhh_b, const float* rlng_f, const float* rlnb_f, const float* rw1_f,
    const float* rb1_f, const float* rw2_f, const float* rb2_f, const float* rlng_b,
    const float* rlnb_b, const float* rw1_b, const float* rb1_b, const float* rw2_b,
    const float* rb2_b) {
  __shared__ float svs[64], uu[64], hh[64], hln[64], tt[128], red2[2];
  int blk = blockIdx.x, tid = threadIdx.x;
  bool isfg = blk < 56;
  int row = isfg ? blk : blk - 56;
  float* slot = (isfg ? slotf : slotb) + row * 64;
  const float* sv  = (isfg ? sfg : sbg) + row * 64;
  const float* mw  = isfg ? mfgw : mbgw;
  const float* mb  = isfg ? mfgb : mbgb;
  const float* Wih = isfg ? wih_f : wih_b;
  const float* Whh = isfg ? whh_f : whh_b;
  const float* Bih = isfg ? bih_f : bih_b;
  const float* Bhh = isfg ? bhh_f : bhh_b;
  const float* lg  = isfg ? rlng_f : rlng_b;
  const float* lb  = isfg ? rlnb_f : rlnb_b;
  const float* w1  = isfg ? rw1_f : rw1_b;
  const float* b1  = isfg ? rb1_f : rb1_b;
  const float* w2  = isfg ? rw2_f : rw2_b;
  const float* b2  = isfg ? rb2_f : rb2_b;

  float h = slot[tid];
  svs[tid] = sv[tid];
  hh[tid] = h;
  __syncthreads();
  // updates = s @ mlpW^T + mlpB  (folded linearity)
  float uo = mb[tid];
  for (int c = 0; c < 64; ++c) uo += svs[c] * mw[tid * 64 + c];
  uu[tid] = uo;
  __syncthreads();
  float gir = Bih[tid], giz = Bih[64 + tid], gin = Bih[128 + tid];
  float ghr = Bhh[tid], ghz = Bhh[64 + tid], ghn = Bhh[128 + tid];
  for (int c = 0; c < 64; ++c) {
    float uc = uu[c], hc = hh[c];
    gir += uc * Wih[tid * 64 + c];
    giz += uc * Wih[(64 + tid) * 64 + c];
    gin += uc * Wih[(128 + tid) * 64 + c];
    ghr += hc * Whh[tid * 64 + c];
    ghz += hc * Whh[(64 + tid) * 64 + c];
    ghn += hc * Whh[(128 + tid) * 64 + c];
  }
  float r  = 1.f / (1.f + expf(-(gir + ghr)));
  float z  = 1.f / (1.f + expf(-(giz + ghz)));
  float nn = tanhf(gin + r * ghn);
  float hnew = (1.f - z) * nn + z * h;
  // residual MLP
  float s  = blockRed64(hnew, red2);
  float s2 = blockRed64(hnew * hnew, red2);
  float m  = s * (1.f / 64.f);
  float inv = rsqrtf(s2 * (1.f / 64.f) - m * m + LNEPS);
  hln[tid] = (hnew - m) * inv * lg[tid] + lb[tid];
  __syncthreads();
  float t0 = b1[tid], t1 = b1[64 + tid];
  for (int c = 0; c < 64; ++c) {
    float hc = hln[c];
    t0 += hc * w1[tid * 64 + c];
    t1 += hc * w1[(64 + tid) * 64 + c];
  }
  tt[tid] = fmaxf(t0, 0.f);
  tt[64 + tid] = fmaxf(t1, 0.f);
  __syncthreads();
  float o = hnew + b2[tid];
  for (int j = 0; j < 128; ++j) o += tt[j] * w2[tid * 128 + j];
  slot[tid] = o;
}

// -------- last-iter texture readout --------
__global__ void tex_accum_kernel(const float* __restrict__ ftf, const float* __restrict__ ftb,
                                 const float* __restrict__ attn, const float* __restrict__ invA,
                                 float* __restrict__ stex) {
  __shared__ float red8[8];
  int blk = blockIdx.x;  // 512 = 64 (b,k) * 8 chunks
  int chunk = blk & 7;
  int bk = blk >> 3;
  int b = bk >> 3, k = bk & 7;
  const float* f = ((k == 0) ? ftb : ftf) + ((size_t)b << 14) * 8;
  const float* at = attn + ((size_t)bk << 14);
  float inva = invA[bk];
  float acc[8] = {0, 0, 0, 0, 0, 0, 0, 0};
  for (int i = 0; i < 8; ++i) {
    int n = chunk * 2048 + i * 256 + threadIdx.x;
    float w = at[n] * inva;
    const float* fr = f + (size_t)n * 8;
#pragma unroll
    for (int t = 0; t < 8; ++t) acc[t] += w * fr[t];
  }
#pragma unroll
  for (int t = 0; t < 8; ++t) {
    float s = blockRed256(acc[t], red8);
    if (threadIdx.x == 0) atomicAdd(&stex[bk * 8 + t], s);
  }
}

// -------- final packing of slots + textures into d_out --------
__global__ void emit_kernel(const float* slotf, const float* slotb, const float* stex,
                            float* __restrict__ out) {
  for (int i = threadIdx.x; i < 4096; i += 512) {
    int b = i >> 9, r = (i >> 6) & 7, c = i & 63;
    out[OUT_SLOTS + i] = (r == 0) ? slotb[b * 64 + c] : slotf[(b * 7 + r - 1) * 64 + c];
  }
  if (threadIdx.x < 512) out[OUT_TEX + threadIdx.x] = stex[threadIdx.x];
}

}  // namespace

extern "C" void kernel_launch(void* const* d_in, const int* in_sizes, int n_in, void* d_out,
                              int out_size, void* d_ws, size_t ws_size, hipStream_t stream) {
  (void)in_sizes; (void)n_in; (void)out_size; (void)ws_size;
  const float* feat = (const float*)d_in[0];
  const float* ftex = (const float*)d_in[1];
  const float* nfg  = (const float*)d_in[2];
  const float* nbg  = (const float*)d_in[3];
  enum {
    GEW = 0, GEB, KFG, VFG, KBG, VBG,
    MFG_LNG, MFG_LNB, MFG_W, MFG_B,
    MBG_LNG, MBG_LNB, MBG_W, MBG_B,
    SMU, SLS, SMUB, SLSB, FPOS,
    QLNG, QLNB, QW, QBLNG, QBLNB, QBW,
    GWIH, GWHH, GBIH, GBHH,
    GBWIH, GBWHH, GBBIH, GBBHH,
    RLNG, RLNB, RW1, RB1, RW2, RB2,
    RBLNG, RBLNB, RBW1, RBB1, RBW2, RBB2,
    FLNG, FLNB,
    TFLNG, TFLNB, TFW, TFB,
    TBLNG, TBLNB, TBW, TBB, NPARAM
  };
  const float* P[NPARAM];
  for (int i = 0; i < NPARAM; ++i) P[i] = (const float*)d_in[4 + i];

  float* ws  = (float*)d_ws;
  float* out = (float*)d_out;
  float* featn  = ws + OFF_FEATN;
  float* E      = ws + OFF_E;
  float* ftf    = ws + OFF_FTF;
  float* ftb    = ws + OFF_FTB;
  float* logits = ws + OFF_LOGITS;
  float* kfb    = ws + OFF_KFB;
  float* vfb    = ws + OFF_VFB;
  float* xkbg   = ws + OFF_XKBG;
  float* xvbg   = ws + OFF_XVBG;
  float* slotf  = ws + OFF_SLOTF;
  float* slotb  = ws + OFF_SLOTB;
  float* pos    = ws + OFF_POS;
  float* wqf    = ws + OFF_WQF;
  float* cf     = ws + OFF_CFC;
  float* cvec   = ws + OFF_CVEC;
  float* wqb    = ws + OFF_WQB;
  float* cb     = ws + OFF_CBC;
  float* Asum   = ws + OFF_ASUM;
  float* Ax     = ws + OFF_AX;
  float* Ay     = ws + OFF_AY;
  float* invA   = ws + OFF_INVA;
  float* sfg    = ws + OFF_SFG;
  float* sbg    = ws + OFF_SBG;
  float* stex   = ws + OFF_STEX;
  float* attn_out = out + OUT_ATTN;
  float* pos_out  = out + OUT_POS;

  // ---- precompute ----
  ln64_kernel<<<BN / 8, 256, 0, stream>>>(feat, P[FLNG], P[FLNB], featn);
  gemm64_wmma_kernel<<<BN / 128, 256, 0, stream>>>(featn, P[KFG], kfb);
  gemm64_wmma_kernel<<<BN / 128, 256, 0, stream>>>(featn, P[VFG], vfb);
  gemm64_wmma_kernel<<<BN / 128, 256, 0, stream>>>(featn, P[KBG], xkbg);
  gemm64_wmma_kernel<<<BN / 128, 256, 0, stream>>>(featn, P[VBG], xvbg);
  gridE_kernel<<<(N * 64) / 256, 256, 0, stream>>>(P[GEW], E);  // reuses featn region
  addE_ln_kernel<<<BN / 8, 256, 0, stream>>>(xkbg, E, P[GEB], P[MBG_LNG], P[MBG_LNB]);
  addE_ln_kernel<<<BN / 8, 256, 0, stream>>>(xvbg, E, P[GEB], P[MBG_LNG], P[MBG_LNB]);
  tex_pre_kernel<<<BN / 256, 256, 0, stream>>>(ftex, P[TFLNG], P[TFLNB], P[TFW], P[TFB],
                                               P[TBLNG], P[TBLNB], P[TBW], P[TBB], ftf, ftb);
  init_kernel<<<14, 256, 0, stream>>>(P[SMU], P[SLS], P[SMUB], P[SLSB], nfg, nbg, P[FPOS],
                                      slotf, slotb, pos);

  // ---- iterations ----
  for (int it = 0; it < 4; ++it) {
    slot_prep_kernel<<<64, 64, 0, stream>>>(slotf, slotb, pos, P[QLNG], P[QLNB], P[QW],
                                            P[QBLNG], P[QBLNB], P[QBW], P[MFG_W], P[MFG_B],
                                            P[MBG_W], P[MBG_B], P[GEW], P[GEB], wqf, cf, cvec,
                                            wqb, cb);
    logits_kernel<<<(NB * 8 * N) / 8, 256, 0, stream>>>(kfb, E, cvec, wqf, cf, xkbg, wqb, cb,
                                                        P[MFG_LNG], P[MFG_LNB], logits);
    zero_kernel<<<1, 256, 0, stream>>>(Asum, 176);  // Asum+Ax+Ay contiguous
    softmax_kernel<<<BN / 256, 256, 0, stream>>>(logits, attn_out, Asum, Ax, Ay);
    post_attn_kernel<<<1, 64, 0, stream>>>(Asum, Ax, Ay, invA, pos, pos_out);
    if (it != 3) {
      zero_kernel<<<16, 256, 0, stream>>>(sfg, 4096);  // sfg+sbg contiguous
      upd_fg_kernel<<<56 * 8, 256, 0, stream>>>(vfb, E, cvec, attn_out, invA, P[MFG_LNG],
                                                P[MFG_LNB], sfg);
      upd_bg_kernel<<<8 * 8, 256, 0, stream>>>(xvbg, attn_out, invA, sbg);
      gru_res_kernel<<<64, 64, 0, stream>>>(
          slotf, slotb, sfg, sbg, P[MFG_W], P[MFG_B], P[MBG_W], P[MBG_B], P[GWIH], P[GWHH],
          P[GBIH], P[GBHH], P[GBWIH], P[GBWHH], P[GBBIH], P[GBBHH], P[RLNG], P[RLNB], P[RW1],
          P[RB1], P[RW2], P[RB2], P[RBLNG], P[RBLNB], P[RBW1], P[RBB1], P[RBW2], P[RBB2]);
    } else {
      zero_kernel<<<2, 256, 0, stream>>>(stex, 512);
      tex_accum_kernel<<<512, 256, 0, stream>>>(ftf, ftb, attn_out, invA, stex);
    }
  }
  emit_kernel<<<1, 512, 0, stream>>>(slotf, slotb, stex, out);
}
